// fasterRCNNFPNBase_79989470921182
// MI455X (gfx1250) — matrix operand order repaired and checked by reading, not compiled, not run on hardware
//
#include <hip/hip_runtime.h>
#include <hip/hip_bf16.h>

// ---------------------------------------------------------------------------
// Faster R-CNN FPN head for MI455X (gfx1250, wave32, WMMA).
// ROI-align -> bf16 WMMA GEMM (12544->1024) -> ReLU -> GEMM (1024->1024)
// -> ReLU -> heads GEMM (1024->{324 loc, 81 cls}) -> softmax.
// GEMMs use v_wmma_f32_16x16x32_bf16 with 2(M) x 4(N) register tiling per
// wave: 8 WMMAs per 6 contiguous 32B fragment loads per k-step, cutting L2
// operand traffic ~2.7x vs one-tile-per-wave.
// ---------------------------------------------------------------------------

typedef __bf16 bf16;
typedef __attribute__((ext_vector_type(16))) __bf16 v16bf;
typedef __attribute__((ext_vector_type(8)))  float  v8f;

static constexpr int N_ROI   = 1000;
static constexpr int MALLOC_ = 1024;   // rows padded to multiple of 32 (2 tiles)
static constexpr int K1      = 12544;  // 256 * 7 * 7
static constexpr int KT1     = 392;    // K1 / 32
static constexpr int NHID    = 1024;
static constexpr int NT_HID  = 64;     // 1024 / 16
static constexpr int KT_HID  = 32;     // 1024 / 32
static constexpr int NHEAD   = 448;    // 324 + 81 padded to 28 tiles
static constexpr int NT_HEAD = 28;
static constexpr int NLOC    = 324;
static constexpr int NCLS    = 81;
static constexpr int LSTRIDE = 96;     // logits row stride

// A-fragment address helper: element (m, k) of a 16x32 bf16 A tile.
// lane = hi2*16 + (m&15); i = rem where k%32 = hi2*16 + rem.
__device__ __forceinline__ long a_frag_index(int m, int k, int Ktiles) {
  int mt  = m >> 4;
  int kt  = k >> 5;
  int kk  = k & 31;
  int rem = kk & 15;
  int hi2 = kk >> 4;
  int i   = rem;
  int lane = hi2 * 16 + (m & 15);
  return (((long)mt * Ktiles + kt) * 32 + lane) * 16 + i;
}

// ---------------------------------------------------------------------------
// Pack fp32 row-major weight [K, N] into WMMA B-fragment bf16 layout:
// frag (kt, nt): lane L holds K = kt*32 + (L>>4)*16 + i, N = nt*16 + (L&15).
// ---------------------------------------------------------------------------
__global__ void pack_w_kernel(const float* __restrict__ src, int N, int Ntiles,
                              bf16* __restrict__ dst, long total) {
  long t = (long)blockIdx.x * blockDim.x + threadIdx.x;
  if (t >= total) return;
  int  lane = (int)(t & 31);
  long frag = t >> 5;
  int  nt   = (int)(frag % Ntiles);
  int  kt   = (int)(frag / Ntiles);
  int  hi   = lane >> 4;
  int  n    = nt * 16 + (lane & 15);
  bf16 tmp[16];
#pragma unroll
  for (int i = 0; i < 16; ++i) {
    int k = kt * 32 + hi * 16 + i;
    tmp[i] = (bf16)src[(long)k * N + n];
  }
  *(v16bf*)(dst + t * 16) = *(const v16bf*)tmp;
}

// Heads weight: logical [1024, 448] = [w_loc | w_cls | zero-pad].
__global__ void pack_heads_kernel(const float* __restrict__ wloc,
                                  const float* __restrict__ wcls,
                                  bf16* __restrict__ dst, long total) {
  long t = (long)blockIdx.x * blockDim.x + threadIdx.x;
  if (t >= total) return;
  int  lane = (int)(t & 31);
  long frag = t >> 5;
  int  nt   = (int)(frag % NT_HEAD);
  int  kt   = (int)(frag / NT_HEAD);
  int  hi   = lane >> 4;
  int  n    = nt * 16 + (lane & 15);
  bf16 tmp[16];
#pragma unroll
  for (int i = 0; i < 16; ++i) {
    int k = kt * 32 + hi * 16 + i;
    float v = 0.0f;
    if (n < NLOC)             v = wloc[(long)k * NLOC + n];
    else if (n < NLOC + NCLS) v = wcls[(long)k * NCLS + (n - NLOC)];
    tmp[i] = (bf16)v;
  }
  *(v16bf*)(dst + t * 16) = *(const v16bf*)tmp;
}

// ---------------------------------------------------------------------------
// ROI-Align: one block (256 threads, channel per thread) per ROI.
// Output written directly as bf16 A-fragments for GEMM1 (k = c*49 + py*7 + px).
// ---------------------------------------------------------------------------
__global__ void roi_align_pack_kernel(const float* __restrict__ p2,
                                      const float* __restrict__ p3,
                                      const float* __restrict__ p4,
                                      const float* __restrict__ p5,
                                      const float* __restrict__ rois,
                                      bf16* __restrict__ Ap,
                                      float* __restrict__ rois_out) {
  int roi = blockIdx.x;
  int tid = threadIdx.x;

  __shared__ const float* sfeat;
  __shared__ int sDim;
  __shared__ int sx0[14], sx1[14], sy0[14], sy1[14];
  __shared__ float swx[14], swy[14];

  const float* rp = rois + (long)roi * 5;
  float x1 = rp[1], y1 = rp[2], x2 = rp[3], y2 = rp[4];

  if (tid < 5) rois_out[(long)roi * 5 + tid] = rp[tid];

  if (tid == 0) {
    float rrw = x2 - x1 + 1.0f;
    float rrh = y2 - y1 + 1.0f;
    float l = rintf(log2f(sqrtf(rrh * rrw) / 224.0f) + 4.0f);
    l = fminf(fmaxf(l, 2.0f), 5.0f);
    int lvl = (int)l;                      // 2..5
    const float* fp[4] = {p2, p3, p4, p5};
    int dims[4] = {200, 100, 50, 25};
    sfeat = fp[lvl - 2];
    sDim  = dims[lvl - 2];
  }
  __syncthreads();

  int D = sDim;                            // H == W
  float scale = (float)D / 800.0f;         // 1/stride for every level

  if (tid < 14) {
    float off = ((float)tid + 0.5f) / 14.0f;
    float xs1 = x1 * scale, xs2 = x2 * scale;
    float rw2 = fmaxf(xs2 - xs1, 1.0f);
    float sx = fminf(fmaxf(xs1 + off * rw2, 0.0f), (float)(D - 1));
    float x0f = floorf(sx);
    sx0[tid] = (int)x0f;
    sx1[tid] = min((int)x0f + 1, D - 1);
    swx[tid] = sx - x0f;

    float ys1 = y1 * scale, ys2 = y2 * scale;
    float rh2 = fmaxf(ys2 - ys1, 1.0f);
    float sy = fminf(fmaxf(ys1 + off * rh2, 0.0f), (float)(D - 1));
    float y0f = floorf(sy);
    sy0[tid] = (int)y0f;
    sy1[tid] = min((int)y0f + 1, D - 1);
    swy[tid] = sy - y0f;
  }
  __syncthreads();

  const float* f = sfeat + (long)tid * D * D;   // this thread's channel
  int m = roi;

#pragma unroll 1
  for (int py = 0; py < 7; ++py) {
#pragma unroll 1
    for (int px = 0; px < 7; ++px) {
      float v = 0.0f;
#pragma unroll
      for (int dy = 0; dy < 2; ++dy) {
        int iy = py * 2 + dy;
        int ya = sy0[iy], yb = sy1[iy];
        float wy = swy[iy];
        const float* r0 = f + (long)ya * D;
        const float* r1 = f + (long)yb * D;
#pragma unroll
        for (int dx = 0; dx < 2; ++dx) {
          int ix = px * 2 + dx;
          int xa = sx0[ix], xb = sx1[ix];
          float wx = swx[ix];
          float v00 = r0[xa], v01 = r0[xb], v10 = r1[xa], v11 = r1[xb];
          v += v00 * (1.0f - wy) * (1.0f - wx) + v01 * (1.0f - wy) * wx +
               v10 * wy * (1.0f - wx)          + v11 * wy * wx;
        }
      }
      v *= 0.25f;
      int k = tid * 49 + py * 7 + px;
      Ap[a_frag_index(m, k, KT1)] = (bf16)v;
    }
  }
}

// ---------------------------------------------------------------------------
// WMMA GEMM with 2(M) x 4(N) register tiling per wave.
// Block = 4 waves = 4 adjacent N-strips. blockIdx.y = M block of 2 tiles.
// mode 0: bias + ReLU, re-pack as bf16 A-fragments for the next layer.
// mode 1: heads -> x_loc straight to d_out, x_cls logits to workspace.
// Buffers are padded so the hot loop needs no bounds checks.
// ---------------------------------------------------------------------------
__global__ void gemm_wmma_tiled(const bf16* __restrict__ Ap,
                                const bf16* __restrict__ Bp,
                                int Ktiles, int NtilesB, int NStrips, int mode,
                                const float* __restrict__ bias0,
                                const float* __restrict__ bias1,
                                bf16* __restrict__ outPack, int KtilesOut,
                                float* __restrict__ outLoc,
                                float* __restrict__ outLogits) {
  int lane  = threadIdx.x & 31;
  int wave  = threadIdx.x >> 5;
  int strip = blockIdx.x * 4 + wave;     // group of 4 N tiles
  int mt0   = blockIdx.y * 2;            // group of 2 M tiles
  if (strip >= NStrips) return;          // no barriers below: wave exit is safe
  int nt0 = strip * 4;

  const v8f zero = {0.f, 0.f, 0.f, 0.f, 0.f, 0.f, 0.f, 0.f};
  v8f acc[2][4];
#pragma unroll
  for (int i = 0; i < 2; ++i)
#pragma unroll
    for (int j = 0; j < 4; ++j) acc[i][j] = zero;

  const v16bf* aP0 = (const v16bf*)Ap + ((long)(mt0 + 0) * Ktiles) * 32 + lane;
  const v16bf* aP1 = (const v16bf*)Ap + ((long)(mt0 + 1) * Ktiles) * 32 + lane;
  const v16bf* bP  = (const v16bf*)Bp + (long)nt0 * 32 + lane;
  const long bStep = (long)NtilesB * 32;

#pragma unroll 1
  for (int kt = 0; kt < Ktiles; ++kt) {
    v16bf a0 = aP0[(long)kt * 32];
    v16bf a1 = aP1[(long)kt * 32];
    v16bf b0 = bP[(long)kt * bStep + 0 * 32];
    v16bf b1 = bP[(long)kt * bStep + 1 * 32];
    v16bf b2 = bP[(long)kt * bStep + 2 * 32];
    v16bf b3 = bP[(long)kt * bStep + 3 * 32];

    // Speculative prefetch of next k-step A fragments (OOB at tail is
    // silently dropped by hardware) -> global_prefetch_b8.
    __builtin_prefetch((const void*)(aP0 + (long)(kt + 1) * 32), 0, 1);
    __builtin_prefetch((const void*)(aP1 + (long)(kt + 1) * 32), 0, 1);

    acc[0][0] = __builtin_amdgcn_wmma_f32_16x16x32_bf16(false, a0, false, b0,
                                                        (short)0, acc[0][0], false, false);
    acc[0][1] = __builtin_amdgcn_wmma_f32_16x16x32_bf16(false, a0, false, b1,
                                                        (short)0, acc[0][1], false, false);
    acc[0][2] = __builtin_amdgcn_wmma_f32_16x16x32_bf16(false, a0, false, b2,
                                                        (short)0, acc[0][2], false, false);
    acc[0][3] = __builtin_amdgcn_wmma_f32_16x16x32_bf16(false, a0, false, b3,
                                                        (short)0, acc[0][3], false, false);
    acc[1][0] = __builtin_amdgcn_wmma_f32_16x16x32_bf16(false, a1, false, b0,
                                                        (short)0, acc[1][0], false, false);
    acc[1][1] = __builtin_amdgcn_wmma_f32_16x16x32_bf16(false, a1, false, b1,
                                                        (short)0, acc[1][1], false, false);
    acc[1][2] = __builtin_amdgcn_wmma_f32_16x16x32_bf16(false, a1, false, b2,
                                                        (short)0, acc[1][2], false, false);
    acc[1][3] = __builtin_amdgcn_wmma_f32_16x16x32_bf16(false, a1, false, b3,
                                                        (short)0, acc[1][3], false, false);
  }

  int hi = lane >> 4;
  int n_local = lane & 15;

  if (mode == 0) {
#pragma unroll
    for (int i = 0; i < 2; ++i) {
      int mt = mt0 + i;
#pragma unroll
      for (int j = 0; j < 4; ++j) {
        int n = (nt0 + j) * 16 + n_local;
        float bias = bias0[n];
#pragma unroll
        for (int r = 0; r < 8; ++r) {
          int m = mt * 16 + r + hi * 8;
          float v = acc[i][j][r] + bias;
          v = v > 0.0f ? v : 0.0f;
          outPack[a_frag_index(m, n, KtilesOut)] = (bf16)v;
        }
      }
    }
  } else {
#pragma unroll
    for (int i = 0; i < 2; ++i) {
      int mt = mt0 + i;
#pragma unroll
      for (int j = 0; j < 4; ++j) {
        int n = (nt0 + j) * 16 + n_local;
#pragma unroll
        for (int r = 0; r < 8; ++r) {
          int m = mt * 16 + r + hi * 8;
          if (m >= N_ROI) continue;
          float v = acc[i][j][r];
          if (n < NLOC) {
            outLoc[(long)m * NLOC + n] = v + bias0[n];
          } else if (n < NLOC + NCLS) {
            outLogits[(long)m * LSTRIDE + (n - NLOC)] = v + bias1[n - NLOC];
          }
        }
      }
    }
  }
}

// ---------------------------------------------------------------------------
// Softmax over 81 classes: one wave per row (wave32 shuffle reductions).
// ---------------------------------------------------------------------------
__global__ void softmax_kernel(const float* __restrict__ logits,
                               float* __restrict__ out) {
  int wave = threadIdx.x >> 5;
  int lane = threadIdx.x & 31;
  int row = blockIdx.x * 8 + wave;
  if (row >= N_ROI) return;
  const float* L = logits + (long)row * LSTRIDE;

  float v0 = (lane      < NCLS) ? L[lane]      : -1e30f;
  float v1 = (lane + 32 < NCLS) ? L[lane + 32] : -1e30f;
  float v2 = (lane + 64 < NCLS) ? L[lane + 64] : -1e30f;

  float mx = fmaxf(v0, fmaxf(v1, v2));
#pragma unroll
  for (int off = 16; off; off >>= 1) mx = fmaxf(mx, __shfl_xor(mx, off, 32));

  float e0 = __expf(v0 - mx);
  float e1 = __expf(v1 - mx);
  float e2 = __expf(v2 - mx);
  float s = e0 + e1 + e2;
#pragma unroll
  for (int off = 16; off; off >>= 1) s += __shfl_xor(s, off, 32);
  float inv = 1.0f / s;

  float* o = out + (long)row * NCLS;
  if (lane      < NCLS) o[lane]      = e0 * inv;
  if (lane + 32 < NCLS) o[lane + 32] = e1 * inv;
  if (lane + 64 < NCLS) o[lane + 64] = e2 * inv;
}

// ---------------------------------------------------------------------------
extern "C" void kernel_launch(void* const* d_in, const int* in_sizes, int n_in,
                              void* d_out, int out_size, void* d_ws, size_t ws_size,
                              hipStream_t stream) {
  const float* p2    = (const float*)d_in[0];
  const float* p3    = (const float*)d_in[1];
  const float* p4    = (const float*)d_in[2];
  const float* p5    = (const float*)d_in[3];
  const float* rois  = (const float*)d_in[4];
  const float* w1    = (const float*)d_in[5];
  const float* b1    = (const float*)d_in[6];
  const float* w2    = (const float*)d_in[7];
  const float* b2    = (const float*)d_in[8];
  const float* wcls  = (const float*)d_in[9];
  const float* bcls  = (const float*)d_in[10];
  const float* wloc  = (const float*)d_in[11];
  const float* bloc  = (const float*)d_in[12];

  float* out      = (float*)d_out;
  float* rois_out = out;                         // [1,1000,5]   = 5000
  float* cls_out  = out + 5000;                  // [1,1000,81]  = 81000
  float* loc_out  = out + 5000 + 81000;          // [1,1000,324] = 324000

  // Workspace carve-up (256B aligned): ~59 MB total.
  char* ws = (char*)d_ws;
  size_t off = 0;
  auto carve = [&](size_t bytes) {
    void* p = ws + off;
    off = (off + bytes + 255) & ~(size_t)255;
    return p;
  };
  bf16*  w1p    = (bf16*)carve((size_t)K1 * NHID * 2);          // 25.7 MB
  bf16*  w2p    = (bf16*)carve((size_t)NHID * NHID * 2);        //  2.1 MB
  bf16*  whp    = (bf16*)carve((size_t)NHID * NHEAD * 2);       //  0.9 MB
  bf16*  Ap     = (bf16*)carve((size_t)MALLOC_ * K1 * 2);       // 25.7 MB
  bf16*  H1p    = (bf16*)carve((size_t)MALLOC_ * NHID * 2);     //  2.1 MB
  bf16*  H2p    = (bf16*)carve((size_t)MALLOC_ * NHID * 2);     //  2.1 MB
  float* logits = (float*)carve((size_t)MALLOC_ * LSTRIDE * 4); //  0.4 MB

  // 1) Pack weights into WMMA B-fragment layouts (bf16).
  long t1 = (long)KT1 * NT_HID * 32;     // 802816 lane-fragments
  pack_w_kernel<<<dim3((unsigned)((t1 + 255) / 256)), 256, 0, stream>>>(
      w1, NHID, NT_HID, w1p, t1);
  long t2 = (long)KT_HID * NT_HID * 32;  // 65536
  pack_w_kernel<<<dim3((unsigned)((t2 + 255) / 256)), 256, 0, stream>>>(
      w2, NHID, NT_HID, w2p, t2);
  long t3 = (long)KT_HID * NT_HEAD * 32; // 28672
  pack_heads_kernel<<<dim3((unsigned)((t3 + 255) / 256)), 256, 0, stream>>>(
      wloc, wcls, whp, t3);

  // 2) ROI-align -> packed bf16 A-fragments, plus rois passthrough.
  roi_align_pack_kernel<<<dim3(N_ROI), 256, 0, stream>>>(
      p2, p3, p4, p5, rois, Ap, rois_out);

  // 3) GEMM1: [1024,12544] x [12544,1024] -> ReLU -> packed H1.
  //    grid: 16 N-strips / 4 waves per block, 32 M-blocks of 2 tiles.
  gemm_wmma_tiled<<<dim3(4, 32), 128, 0, stream>>>(
      Ap, w1p, KT1, NT_HID, /*NStrips=*/16, 0, b1, nullptr,
      H1p, KT_HID, nullptr, nullptr);

  // 4) GEMM2: [1024,1024] x [1024,1024] -> ReLU -> packed H2.
  gemm_wmma_tiled<<<dim3(4, 32), 128, 0, stream>>>(
      H1p, w2p, KT_HID, NT_HID, /*NStrips=*/16, 0, b2, nullptr,
      H2p, KT_HID, nullptr, nullptr);

  // 5) Heads: [1024,1024] x [1024,448] -> x_loc to d_out, logits to ws.
  //    7 N-strips (28 tiles), guard inside kernel.
  gemm_wmma_tiled<<<dim3(2, 32), 128, 0, stream>>>(
      H2p, whp, KT_HID, NT_HEAD, /*NStrips=*/7, 1, bloc, bcls,
      nullptr, 0, loc_out, logits);

  // 6) Softmax over 81 classes.
  softmax_kernel<<<dim3((N_ROI + 7) / 8), 256, 0, stream>>>(logits, cls_out);
}
